// MoEUpProjWithLoRA_2336462209575
// MI455X (gfx1250) — compile-verified, matrix operand order checked by reading, not run on hardware
//
#include <hip/hip_runtime.h>
#include <stdint.h>

#define HDIM 4096
#define NEXP 4
#define RANK 8
#define LSCALE 1.0f   // ALPHA / RANK

#define BM 128
#define BN 256
#define BK 64
#define LDK 72        // padded LDS row stride in bf16 elems (multiple of 8 -> 16B aligned b128)

typedef __attribute__((ext_vector_type(16))) __bf16 v16bf;
typedef __attribute__((ext_vector_type(8)))  __bf16 v8bf;
typedef __attribute__((ext_vector_type(8)))  float  v8f;
typedef int v4i __attribute__((vector_size(16)));   // matches async-LDS builtin param type

#if defined(__has_builtin)
#if __has_builtin(__builtin_amdgcn_global_load_async_to_lds_b128)
#define HAVE_ASYNC_LDS 1
#endif
#endif

__device__ __forceinline__ uint32_t pk2bf(float a, float b) {
  union { float f; uint32_t u; } x, y;
  x.f = a; y.f = b;
  uint32_t lo = (x.u + 0x7FFFu + ((x.u >> 16) & 1u)) >> 16;   // RNE f32->bf16
  uint32_t hi = (y.u + 0x7FFFu + ((y.u >> 16) & 1u)) >> 16;
  return lo | (hi << 16);
}

// 16-byte global->LDS copy: async (ASYNCcnt) when available, else load+ds_store.
__device__ __forceinline__ void cp_b128(const void* g, void* l) {
#if defined(HAVE_ASYNC_LDS)
  __builtin_amdgcn_global_load_async_to_lds_b128(
      (__attribute__((address_space(1))) v4i*)g,
      (__attribute__((address_space(3))) v4i*)l, 0, 0);
#else
  *reinterpret_cast<uint4*>(l) = *reinterpret_cast<const uint4*>(g);
#endif
}

__device__ __forceinline__ void wait_async_copies() {
#if defined(HAVE_ASYNC_LDS)
#if __has_builtin(__builtin_amdgcn_s_wait_asynccnt)
  __builtin_amdgcn_s_wait_asynccnt(0);
#else
  asm volatile("s_wait_asynccnt 0" ::: "memory");
#endif
#endif
}

// ---------------------------------------------------------------------------
// Kernel 0: f32 -> bf16 pre-convert (one-time, streaming)
// ---------------------------------------------------------------------------
__global__ __launch_bounds__(256) void cvt_bf16_kernel(
    const float* __restrict__ src, uint16_t* __restrict__ dst)
{
  const size_t i = (size_t)blockIdx.x * 256 + threadIdx.x;  // float4 index
  const float4 v = reinterpret_cast<const float4*>(src)[i];
  reinterpret_cast<uint2*>(dst)[i] = make_uint2(pk2bf(v.x, v.y), pk2bf(v.z, v.w));
}

// ---------------------------------------------------------------------------
// Kernel 1: top-1 gating + LoRA "z" (z[n,r] = sum_h x[n,h] * lora_A[e,r,h])
// ---------------------------------------------------------------------------
__global__ __launch_bounds__(256) void moe_gate_z_kernel(
    const float* __restrict__ x, const float* __restrict__ gate_w,
    const float* __restrict__ ebias, const float* __restrict__ lora_A,
    int* __restrict__ eidx, float* __restrict__ z)
{
  const int wid  = threadIdx.x >> 5;
  const int lane = threadIdx.x & 31;
  const int n    = blockIdx.x * 8 + wid;

  const float4* xrow = reinterpret_cast<const float4*>(x + (size_t)n * HDIM);
  float4 xv[32];
  #pragma unroll
  for (int i = 0; i < 32; ++i) xv[i] = xrow[i * 32 + lane];

  float acc[NEXP] = {0.f, 0.f, 0.f, 0.f};
  #pragma unroll
  for (int i = 0; i < 32; ++i) {
    #pragma unroll
    for (int e = 0; e < NEXP; ++e) {
      float4 g = reinterpret_cast<const float4*>(gate_w + (size_t)e * HDIM)[i * 32 + lane];
      acc[e] += xv[i].x * g.x + xv[i].y * g.y + xv[i].z * g.z + xv[i].w * g.w;
    }
  }
  #pragma unroll
  for (int e = 0; e < NEXP; ++e) {
    #pragma unroll
    for (int s = 16; s > 0; s >>= 1) acc[e] += __shfl_xor(acc[e], s, 32);
  }
  int best = 0;
  float bv = acc[0] + ebias[0];
  #pragma unroll
  for (int e = 1; e < NEXP; ++e) {
    float v = acc[e] + ebias[e];
    if (v > bv) { bv = v; best = e; }
  }

  float zz[RANK];
  #pragma unroll
  for (int r = 0; r < RANK; ++r) zz[r] = 0.f;
  const float* Abase = lora_A + (size_t)best * RANK * HDIM;
  #pragma unroll
  for (int i = 0; i < 32; ++i) {
    #pragma unroll
    for (int r = 0; r < RANK; ++r) {
      float4 a = reinterpret_cast<const float4*>(Abase + (size_t)r * HDIM)[i * 32 + lane];
      zz[r] += xv[i].x * a.x + xv[i].y * a.y + xv[i].z * a.z + xv[i].w * a.w;
    }
  }
  #pragma unroll
  for (int r = 0; r < RANK; ++r) {
    #pragma unroll
    for (int s = 16; s > 0; s >>= 1) zz[r] += __shfl_xor(zz[r], s, 32);
  }
  if (lane == 0) {
    eidx[n] = best;
    #pragma unroll
    for (int r = 0; r < RANK; ++r) z[(size_t)n * RANK + r] = zz[r] * LSCALE;
  }
}

// ---------------------------------------------------------------------------
// Primary GEMM: pre-converted bf16 inputs, double-buffered LDS, async staging
// ---------------------------------------------------------------------------
union MoeSmem2 {
  struct { uint16_t A[BM][LDK]; uint16_t B[BN][LDK]; } g[2];              // 110592 B
  struct { float z[BM][RANK]; float lb[NEXP][BN][RANK]; int e[BM]; } ep;  //  37376 B
};

__device__ __forceinline__ void stage_tiles_bf16(
    MoeSmem2* sm, int buf, const uint16_t* __restrict__ xb,
    const uint16_t* __restrict__ wb, int m0, int n0, int k0, int t)
{
  #pragma unroll
  for (int it = 0; it < 4; ++it) {          // A: 128x64 bf16 = 1024 x 16B
    int idx = it * 256 + t;
    int row = idx >> 3, c8 = (idx & 7) << 3;
    cp_b128(xb + (size_t)(m0 + row) * HDIM + k0 + c8, &sm->g[buf].A[row][c8]);
  }
  #pragma unroll
  for (int it = 0; it < 8; ++it) {          // B: 256x64 bf16 = 2048 x 16B
    int idx = it * 256 + t;
    int row = idx >> 3, c8 = (idx & 7) << 3;
    cp_b128(wb + (size_t)(n0 + row) * HDIM + k0 + c8, &sm->g[buf].B[row][c8]);
  }
}

__global__ __launch_bounds__(256) void moe_wmma_gemm_bf16_kernel(
    const uint16_t* __restrict__ xb, const uint16_t* __restrict__ wb,
    const float* __restrict__ lora_B, const int* __restrict__ eidx,
    const float* __restrict__ z, float* __restrict__ out)
{
  __shared__ MoeSmem2 sm;

  const int t    = threadIdx.x;
  const int lane = t & 31;
  const int wid  = t >> 5;
  const int wm   = wid >> 2;
  const int wn   = wid & 3;
  const int m0   = blockIdx.y * BM;
  const int n0   = blockIdx.x * BN;
  const int half = (lane < 16) ? 0 : 8;
  const int r16  = lane & 15;

  v8f acc[4][4] = {};

  stage_tiles_bf16(&sm, 0, xb, wb, m0, n0, 0, t);
  int cur = 0;
  for (int k0 = 0; k0 < HDIM; k0 += BK) {
    wait_async_copies();
    __syncthreads();
    if (k0 + BK < HDIM)
      stage_tiles_bf16(&sm, cur ^ 1, xb, wb, m0, n0, k0 + BK, t);  // overlaps WMMA below

    #pragma unroll
    for (int ks = 0; ks < 2; ++ks) {
      const int kb = ks * 32 + half;
      v16bf a[4];
      #pragma unroll
      for (int mi = 0; mi < 4; ++mi) {
        const uint16_t* rp = &sm.g[cur].A[wm * 64 + mi * 16 + r16][0];
        v8bf lo = *reinterpret_cast<const v8bf*>(rp + kb);
        v8bf hi = *reinterpret_cast<const v8bf*>(rp + kb + 16);
        a[mi] = __builtin_shufflevector(lo, hi, 0,1,2,3,4,5,6,7,8,9,10,11,12,13,14,15);
      }
      #pragma unroll
      for (int ni = 0; ni < 4; ++ni) {
        const uint16_t* rp = &sm.g[cur].B[wn * 64 + ni * 16 + r16][0];
        v8bf lo = *reinterpret_cast<const v8bf*>(rp + kb);
        v8bf hi = *reinterpret_cast<const v8bf*>(rp + kb + 16);
        v16bf b = __builtin_shufflevector(lo, hi, 0,1,2,3,4,5,6,7,8,9,10,11,12,13,14,15);
        #pragma unroll
        for (int mi = 0; mi < 4; ++mi)
          acc[mi][ni] = __builtin_amdgcn_wmma_f32_16x16x32_bf16(
              false, a[mi], false, b, (short)0, acc[mi][ni], false, false);
      }
    }
    cur ^= 1;
  }

  // ---- epilogue: fused rank-8 LoRA delta ----
  __syncthreads();
  if (t < BM) {
    const float4* zp = reinterpret_cast<const float4*>(z + (size_t)(m0 + t) * RANK);
    *reinterpret_cast<float4*>(&sm.ep.z[t][0]) = zp[0];
    *reinterpret_cast<float4*>(&sm.ep.z[t][4]) = zp[1];
    sm.ep.e[t] = eidx[m0 + t];
  }
  #pragma unroll
  for (int it = 0; it < 8; ++it) {            // lora_B: 4 * 256 * 32B, raw copy
    int idx = it * 256 + t;
    int e = idx >> 9, rem = idx & 511;
    int nn = rem >> 1, p = (rem & 1) << 2;
    cp_b128(lora_B + ((size_t)e * HDIM + n0 + nn) * RANK + p, &sm.ep.lb[e][nn][p]);
  }
  wait_async_copies();
  __syncthreads();

  #pragma unroll
  for (int mi = 0; mi < 4; ++mi) {
    #pragma unroll
    for (int i = 0; i < 8; ++i) {
      const int ml = wm * 64 + mi * 16 + half + i;
      const int e  = sm.ep.e[ml];
      const float4 z0 = *reinterpret_cast<const float4*>(&sm.ep.z[ml][0]);
      const float4 z1 = *reinterpret_cast<const float4*>(&sm.ep.z[ml][4]);
      #pragma unroll
      for (int ni = 0; ni < 4; ++ni) {
        const int nl = wn * 64 + ni * 16 + r16;
        const float4 b0 = *reinterpret_cast<const float4*>(&sm.ep.lb[e][nl][0]);
        const float4 b1 = *reinterpret_cast<const float4*>(&sm.ep.lb[e][nl][4]);
        const float d = z0.x * b0.x + z0.y * b0.y + z0.z * b0.z + z0.w * b0.w
                      + z1.x * b1.x + z1.y * b1.y + z1.z * b1.z + z1.w * b1.w;
        out[(size_t)(m0 + ml) * HDIM + n0 + nl] = acc[mi][ni][i] + d;
      }
    }
  }
}

// ---------------------------------------------------------------------------
// Fallback GEMM: fp32 inputs, convert during LDS staging (works with tiny ws)
// ---------------------------------------------------------------------------
union MoeSmem {
  struct { uint16_t A[BM][LDK]; uint16_t B[BN][LDK]; } g;
  struct { float z[BM][RANK]; float lb[NEXP][BN][RANK]; int e[BM]; } ep;
};

__global__ __launch_bounds__(256) void moe_wmma_gemm_kernel(
    const float* __restrict__ x, const float* __restrict__ Wup,
    const float* __restrict__ lora_B, const int* __restrict__ eidx,
    const float* __restrict__ z, float* __restrict__ out)
{
  __shared__ MoeSmem sm;

  const int t    = threadIdx.x;
  const int lane = t & 31;
  const int wid  = t >> 5;
  const int wm   = wid >> 2;
  const int wn   = wid & 3;
  const int m0   = blockIdx.y * BM;
  const int n0   = blockIdx.x * BN;
  const int half = (lane < 16) ? 0 : 8;
  const int r16  = lane & 15;

  v8f acc[4][4] = {};

  for (int k0 = 0; k0 < HDIM; k0 += BK) {
    __syncthreads();
    #pragma unroll 4
    for (int it = 0; it < 8; ++it) {
      int idx = it * 256 + t;
      int row = idx >> 4, c4 = (idx & 15) << 2;
      const float4 v = *reinterpret_cast<const float4*>(x + (size_t)(m0 + row) * HDIM + k0 + c4);
      *reinterpret_cast<uint2*>(&sm.g.A[row][c4]) = make_uint2(pk2bf(v.x, v.y), pk2bf(v.z, v.w));
    }
    #pragma unroll 4
    for (int it = 0; it < 16; ++it) {
      int idx = it * 256 + t;
      int row = idx >> 4, c4 = (idx & 15) << 2;
      const float4 v = *reinterpret_cast<const float4*>(Wup + (size_t)(n0 + row) * HDIM + k0 + c4);
      *reinterpret_cast<uint2*>(&sm.g.B[row][c4]) = make_uint2(pk2bf(v.x, v.y), pk2bf(v.z, v.w));
    }
    __syncthreads();

    #pragma unroll
    for (int ks = 0; ks < 2; ++ks) {
      const int kb = ks * 32 + half;
      v16bf a[4];
      #pragma unroll
      for (int mi = 0; mi < 4; ++mi) {
        const uint16_t* rp = &sm.g.A[wm * 64 + mi * 16 + r16][0];
        v8bf lo = *reinterpret_cast<const v8bf*>(rp + kb);
        v8bf hi = *reinterpret_cast<const v8bf*>(rp + kb + 16);
        a[mi] = __builtin_shufflevector(lo, hi, 0,1,2,3,4,5,6,7,8,9,10,11,12,13,14,15);
      }
      #pragma unroll
      for (int ni = 0; ni < 4; ++ni) {
        const uint16_t* rp = &sm.g.B[wn * 64 + ni * 16 + r16][0];
        v8bf lo = *reinterpret_cast<const v8bf*>(rp + kb);
        v8bf hi = *reinterpret_cast<const v8bf*>(rp + kb + 16);
        v16bf b = __builtin_shufflevector(lo, hi, 0,1,2,3,4,5,6,7,8,9,10,11,12,13,14,15);
        #pragma unroll
        for (int mi = 0; mi < 4; ++mi)
          acc[mi][ni] = __builtin_amdgcn_wmma_f32_16x16x32_bf16(
              false, a[mi], false, b, (short)0, acc[mi][ni], false, false);
      }
    }
  }

  __syncthreads();
  if (t < BM) {
    const float4* zp = reinterpret_cast<const float4*>(z + (size_t)(m0 + t) * RANK);
    *reinterpret_cast<float4*>(&sm.ep.z[t][0]) = zp[0];
    *reinterpret_cast<float4*>(&sm.ep.z[t][4]) = zp[1];
    sm.ep.e[t] = eidx[m0 + t];
  }
  #pragma unroll 4
  for (int it = 0; it < 8; ++it) {
    int idx = it * 256 + t;
    int e = idx >> 9, rem = idx & 511;
    int nn = rem >> 1, p = (rem & 1) << 2;
    const float4 v = *reinterpret_cast<const float4*>(
        lora_B + ((size_t)e * HDIM + n0 + nn) * RANK + p);
    *reinterpret_cast<float4*>(&sm.ep.lb[e][nn][p]) = v;
  }
  __syncthreads();

  #pragma unroll
  for (int mi = 0; mi < 4; ++mi) {
    #pragma unroll
    for (int i = 0; i < 8; ++i) {
      const int ml = wm * 64 + mi * 16 + half + i;
      const int e  = sm.ep.e[ml];
      const float4 z0 = *reinterpret_cast<const float4*>(&sm.ep.z[ml][0]);
      const float4 z1 = *reinterpret_cast<const float4*>(&sm.ep.z[ml][4]);
      #pragma unroll
      for (int ni = 0; ni < 4; ++ni) {
        const int nl = wn * 64 + ni * 16 + r16;
        const float4 b0 = *reinterpret_cast<const float4*>(&sm.ep.lb[e][nl][0]);
        const float4 b1 = *reinterpret_cast<const float4*>(&sm.ep.lb[e][nl][4]);
        const float d = z0.x * b0.x + z0.y * b0.y + z0.z * b0.z + z0.w * b0.w
                      + z1.x * b1.x + z1.y * b1.y + z1.z * b1.z + z1.w * b1.w;
        out[(size_t)(m0 + ml) * HDIM + n0 + nl] = acc[mi][ni][i] + d;
      }
    }
  }
}

// ---------------------------------------------------------------------------
extern "C" void kernel_launch(void* const* d_in, const int* in_sizes, int n_in,
                              void* d_out, int out_size, void* d_ws, size_t ws_size,
                              hipStream_t stream) {
  const float* x      = (const float*)d_in[0];
  const float* Wup    = (const float*)d_in[1];
  const float* gate_w = (const float*)d_in[2];
  const float* ebias  = (const float*)d_in[3];
  const float* lora_A = (const float*)d_in[4];
  const float* lora_B = (const float*)d_in[5];
  float* out = (float*)d_out;

  const int Ntok = in_sizes[0] / HDIM;  // 16384

  auto align256 = [](size_t v) { return (v + 255) & ~(size_t)255; };
  size_t off = 0;
  int*   eidx = (int*)((char*)d_ws + off);    off += align256((size_t)Ntok * sizeof(int));
  float* zbuf = (float*)((char*)d_ws + off);  off += align256((size_t)Ntok * RANK * sizeof(float));
  uint16_t* xb = (uint16_t*)((char*)d_ws + off); off += align256((size_t)Ntok * HDIM * 2);
  uint16_t* wb = (uint16_t*)((char*)d_ws + off); off += (size_t)HDIM * HDIM * 2;
  const bool big_ws = (ws_size >= off);

  moe_gate_z_kernel<<<Ntok / 8, 256, 0, stream>>>(x, gate_w, ebias, lora_A, eidx, zbuf);

  dim3 grid(HDIM / BN, Ntok / BM);
  if (big_ws) {
    cvt_bf16_kernel<<<(unsigned)((size_t)Ntok * HDIM / 4 / 256), 256, 0, stream>>>(x, xb);
    cvt_bf16_kernel<<<(unsigned)((size_t)HDIM * HDIM / 4 / 256), 256, 0, stream>>>(Wup, wb);
    moe_wmma_gemm_bf16_kernel<<<grid, 256, 0, stream>>>(xb, wb, lora_B, eidx, zbuf, out);
  } else {
    moe_wmma_gemm_kernel<<<grid, 256, 0, stream>>>(x, Wup, lora_B, eidx, zbuf, out);
  }
}
